// MambaModule_19980187861552
// MI455X (gfx1250) — compile-verified, matrix operand order checked by reading, not run on hardware
//
#include <hip/hip_runtime.h>
#include <hip/hip_bf16.h>

#define BATCH 2
#define SEQ   1024
#define DMOD  1024
#define DIN   2048
#define NXZ   4096
#define MTOT  (BATCH*SEQ)
#define DTR   64
#define DSTATE 16

typedef __attribute__((ext_vector_type(16))) __bf16 v16bf;
typedef __attribute__((ext_vector_type(8)))  float  v8f;
typedef __attribute__((ext_vector_type(4)))  unsigned int u32x4;

union Frag { v16bf v; u32x4 q[2]; };

__device__ __forceinline__ __bf16 f2bf(float f) {
    union { float f; unsigned u; } v; v.f = f;
    unsigned r = v.u + 0x7FFFu + ((v.u >> 16) & 1u);
    unsigned short h = (unsigned short)(r >> 16);
    return __builtin_bit_cast(__bf16, h);
}
__device__ __forceinline__ float bf2f(__bf16 b) {
    unsigned short h = __builtin_bit_cast(unsigned short, b);
    union { unsigned u; float f; } v; v.u = ((unsigned)h) << 16;
    return v.f;
}
__device__ __forceinline__ float sigm(float x) { return 1.0f / (1.0f + __expf(-x)); }

// ---------------------------------------------------------------- f32 -> bf16
__global__ void cvt_bf16_kernel(const float* __restrict__ src,
                                __bf16* __restrict__ dst, long n) {
    long i = (long)blockIdx.x * blockDim.x + threadIdx.x;
    if (i < n) dst[i] = f2bf(src[i]);
}

// ------------------------------------------------- WMMA bf16 GEMM, C = A*B^T
// A: (M,K) bf16 row-major lda. Bm: (N,K) bf16 row-major ldb (weights).
// Per ISA: 16-bit A 16x32 fragment -> lanes 0..15 hold M=lane, K in
// {0..7}U{16..23}; lanes 16..31 hold K in {8..15}U{24..31}. B fragment is the
// mirror with N in place of M, so both load identically from row-major (*,K).
// One wave per block computes a 64(M) x 32(N) C tile: 8 accumulators (64 VGPR)
// + 6 operand fragments (48 VGPR) -> no spills.
__device__ __forceinline__ void load_frag(Frag& f, const __bf16* __restrict__ base,
                                          long row, int ld, int k0, int g) {
    const __bf16* p = base + row * (long)ld + k0;
    f.q[0] = *(const u32x4*)(p + g);        // 8 bf16, K = g..g+7
    f.q[1] = *(const u32x4*)(p + 16 + g);   // 8 bf16, K = 16+g..16+g+7
}

__global__ __launch_bounds__(32, 1)
void gemm_bf16_wmma(const __bf16* __restrict__ A, int lda,
                    const __bf16* __restrict__ Bm, int ldb,
                    int M, int N, int K,
                    float* __restrict__ Cf, int ldc,
                    __bf16* __restrict__ Cb, int ldcb, int bfcols,
                    const float* __restrict__ bias) {
    const int lane = threadIdx.x & 31;
    const int m0 = blockIdx.y * 64;
    const int n0 = blockIdx.x * 32;
    const int g = (lane & 16) ? 8 : 0;
    const int r15 = lane & 15;

    v8f c[4][2];
#pragma unroll
    for (int i = 0; i < 4; ++i)
#pragma unroll
        for (int j = 0; j < 2; ++j) c[i][j] = (v8f)0.0f;

    for (int k0 = 0; k0 < K; k0 += 32) {
        Frag a[4], b[2];
#pragma unroll
        for (int i = 0; i < 4; ++i)
            load_frag(a[i], A, (long)(m0 + 16 * i + r15), lda, k0, g);
#pragma unroll
        for (int j = 0; j < 2; ++j) {
            int r = n0 + 16 * j + r15;
            if (r > N - 1) r = N - 1;
            load_frag(b[j], Bm, (long)r, ldb, k0, g);
        }
#pragma unroll
        for (int i = 0; i < 4; ++i)
#pragma unroll
            for (int j = 0; j < 2; ++j)
                c[i][j] = __builtin_amdgcn_wmma_f32_16x16x32_bf16(
                    false, a[i].v, false, b[j].v, (short)0, c[i][j],
                    false, false);
    }

    // C/D layout: N = lane&15 ; M = ((lane&16)?8:0)+v within each 16x16 tile
    const int mlo = (lane & 16) ? 8 : 0;
#pragma unroll
    for (int j = 0; j < 2; ++j) {
        int nn = n0 + 16 * j + r15;
        if (nn >= N) continue;
        float bv = bias ? bias[nn] : 0.0f;
#pragma unroll
        for (int i = 0; i < 4; ++i) {
            int mb = m0 + 16 * i + mlo;
#pragma unroll
            for (int v = 0; v < 8; ++v) {
                float val = c[i][j][v] + bv;
                long mm = mb + v;
                if (Cf) Cf[mm * (long)ldc + nn] = val;
                if (Cb && nn < bfcols) Cb[mm * (long)ldcb + nn] = f2bf(val);
            }
        }
    }
}

// --------------------------------------- causal depthwise conv (k=4) + SiLU
__global__ void dwconv_silu_kernel(const float* __restrict__ xz,
                                   const float* __restrict__ w,
                                   const float* __restrict__ bias,
                                   __bf16* __restrict__ u_bf, int flip) {
    long idx = (long)blockIdx.x * blockDim.x + threadIdx.x;
    if (idx >= (long)MTOT * DIN) return;
    int c = (int)(idx % DIN);
    int m = (int)(idx / DIN);
    int b = m / SEQ, l = m % SEQ;
    float acc = bias[c];
#pragma unroll
    for (int j = 0; j < 4; ++j) {
        int ls = l - 3 + j;
        if (ls < 0) continue;
        int src = flip ? (SEQ - 1 - ls) : ls;
        acc += xz[((long)(b * SEQ + src)) * NXZ + c] * w[c * 4 + j];
    }
    float y = acc * sigm(acc);
    u_bf[(long)m * DIN + c] = f2bf(y);
}

// ----------------------------------------------- selective scan (recurrence)
// one 16-lane group per (batch, channel); lane n holds state n of 16.
__global__ void selective_scan_kernel(const __bf16* __restrict__ u_bf,
                                      const __bf16* __restrict__ delta_bf,
                                      const float* __restrict__ proj,
                                      const float* __restrict__ A_log,
                                      const float* __restrict__ Dvec,
                                      const float* __restrict__ xz,
                                      float* __restrict__ outz, int flip) {
    const int lane = threadIdx.x & 31;
    const int wave = blockIdx.x * (blockDim.x >> 5) + (threadIdx.x >> 5);
    const int sub = lane >> 4;
    const int n = lane & 15;
    const int pair = wave * 2 + sub;           // (b,d) pair, exact fit
    if (pair >= BATCH * DIN) return;
    const int b = pair / DIN;
    const int d = pair % DIN;

    const float A = -__expf(A_log[d * DSTATE + n]);
    const float Dd = Dvec[d];
    float state = 0.0f;

    for (int l = 0; l < SEQ; ++l) {
        long m = (long)b * SEQ + l;
        float delta = bf2f(delta_bf[m * DIN + d]);
        delta = (delta > 20.0f) ? delta : log1pf(__expf(delta)); // softplus
        float uu = bf2f(u_bf[m * DIN + d]);
        float Bn = proj[m * 96 + DTR + n];
        float Cn = proj[m * 96 + DTR + DSTATE + n];
        state = state * __expf(delta * A) + delta * uu * Bn;
        float part = state * Cn;
        part += __shfl_xor(part, 8, 16);
        part += __shfl_xor(part, 4, 16);
        part += __shfl_xor(part, 2, 16);
        part += __shfl_xor(part, 1, 16);
        if (n == 0) {
            int lo = flip ? (SEQ - 1 - l) : l;
            long mo = (long)b * SEQ + lo;
            float z = xz[mo * NXZ + DIN + d];
            float val = (part + uu * Dd) * (z * sigm(z)) * 0.5f;
            if (flip) outz[mo * DIN + d] += val;
            else      outz[mo * DIN + d] = val;
        }
    }
}

// ---------------------------------------------------------------------- host
static inline unsigned ceil_div(long a, long b) { return (unsigned)((a + b - 1) / b); }

extern "C" void kernel_launch(void* const* d_in, const int* in_sizes, int n_in,
                              void* d_out, int out_size, void* d_ws, size_t ws_size,
                              hipStream_t stream) {
    (void)in_sizes; (void)n_in; (void)out_size; (void)ws_size;
    const float* hidden  = (const float*)d_in[0];
    const float* W_in    = (const float*)d_in[1];
    const float* conv_w  = (const float*)d_in[2];
    const float* conv_b  = (const float*)d_in[3];
    const float* conv_wb = (const float*)d_in[4];
    const float* conv_bb = (const float*)d_in[5];
    const float* W_x     = (const float*)d_in[6];
    const float* W_dt    = (const float*)d_in[7];
    const float* b_dt    = (const float*)d_in[8];
    const float* W_x_b   = (const float*)d_in[9];
    const float* W_dt_b  = (const float*)d_in[10];
    const float* b_dt_b  = (const float*)d_in[11];
    const float* A_log   = (const float*)d_in[12];
    const float* A_b_log = (const float*)d_in[13];
    const float* Dv      = (const float*)d_in[14];
    const float* Dv_b    = (const float*)d_in[15];
    const float* W_out   = (const float*)d_in[16];
    float* out = (float*)d_out;

    char* ws = (char*)d_ws;
    size_t off = 0;
    auto carve = [&](size_t bytes) {
        void* p = ws + off;
        off = (off + bytes + 255) & ~(size_t)255;
        return p;
    };
    __bf16* hbf       = (__bf16*)carve((size_t)MTOT * DMOD * 2);
    __bf16* Win_bf    = (__bf16*)carve((size_t)NXZ * DMOD * 2);
    __bf16* Wx_bf     = (__bf16*)carve((size_t)96 * DIN * 2);
    __bf16* Wxb_bf    = (__bf16*)carve((size_t)96 * DIN * 2);
    __bf16* Wdt_bf    = (__bf16*)carve((size_t)DIN * DTR * 2);
    __bf16* Wdtb_bf   = (__bf16*)carve((size_t)DIN * DTR * 2);
    __bf16* Wout_bf   = (__bf16*)carve((size_t)DMOD * DIN * 2);
    float*  xz        = (float*)carve((size_t)MTOT * NXZ * 4);
    __bf16* uf_bf     = (__bf16*)carve((size_t)MTOT * DIN * 2);
    __bf16* ub_bf     = (__bf16*)carve((size_t)MTOT * DIN * 2);
    float*  proj_f    = (float*)carve((size_t)MTOT * 96 * 4);
    float*  proj_b    = (float*)carve((size_t)MTOT * 96 * 4);
    __bf16* pdtf_bf   = (__bf16*)carve((size_t)MTOT * DTR * 2);
    __bf16* pdtb_bf   = (__bf16*)carve((size_t)MTOT * DTR * 2);
    __bf16* deltaf_bf = (__bf16*)carve((size_t)MTOT * DIN * 2);
    __bf16* deltab_bf = (__bf16*)carve((size_t)MTOT * DIN * 2);
    float*  outz      = (float*)carve((size_t)MTOT * DIN * 4);
    __bf16* outz_bf   = (__bf16*)carve((size_t)MTOT * DIN * 2);

    auto cvt = [&](const float* s, __bf16* d, long n) {
        cvt_bf16_kernel<<<ceil_div(n, 256), 256, 0, stream>>>(s, d, n);
    };
    auto gemm = [&](const __bf16* A, int lda, const __bf16* B, int ldb,
                    int M, int N, int K, float* Cf, int ldc,
                    __bf16* Cb, int ldcb, int bfcols, const float* bias) {
        dim3 grid(ceil_div(N, 32), ceil_div(M, 64));
        gemm_bf16_wmma<<<grid, 32, 0, stream>>>(A, lda, B, ldb, M, N, K,
                                                Cf, ldc, Cb, ldcb, bfcols, bias);
    };

    // stage bf16 operands
    cvt(hidden, hbf, (long)MTOT * DMOD);
    cvt(W_in, Win_bf, (long)NXZ * DMOD);
    cvt(W_x, Wx_bf, (long)96 * DIN);
    cvt(W_x_b, Wxb_bf, (long)96 * DIN);
    cvt(W_dt, Wdt_bf, (long)DIN * DTR);
    cvt(W_dt_b, Wdtb_bf, (long)DIN * DTR);
    cvt(W_out, Wout_bf, (long)DMOD * DIN);

    // xz = hidden @ W_in^T
    gemm(hbf, DMOD, Win_bf, DMOD, MTOT, NXZ, DMOD, xz, NXZ,
         nullptr, 0, 0, nullptr);

    // depthwise conv + SiLU (forward / flipped backward)
    {
        unsigned g = ceil_div((long)MTOT * DIN, 256);
        dwconv_silu_kernel<<<g, 256, 0, stream>>>(xz, conv_w, conv_b, uf_bf, 0);
        dwconv_silu_kernel<<<g, 256, 0, stream>>>(xz, conv_wb, conv_bb, ub_bf, 1);
    }

    // proj = u @ W_x^T  (f32 for B/C cols; bf16 copy of first 64 cols for dt)
    gemm(uf_bf, DIN, Wx_bf, DIN, MTOT, 96, DIN, proj_f, 96,
         pdtf_bf, DTR, DTR, nullptr);
    gemm(ub_bf, DIN, Wxb_bf, DIN, MTOT, 96, DIN, proj_b, 96,
         pdtb_bf, DTR, DTR, nullptr);

    // delta = proj[:, :64] @ W_dt^T + b_dt  (bf16 output)
    gemm(pdtf_bf, DTR, Wdt_bf, DTR, MTOT, DIN, DTR, nullptr, 0,
         deltaf_bf, DIN, DIN, b_dt);
    gemm(pdtb_bf, DTR, Wdtb_bf, DTR, MTOT, DIN, DTR, nullptr, 0,
         deltab_bf, DIN, DIN, b_dt_b);

    // selective scans; forward writes, backward accumulates at flipped rows
    {
        unsigned nblocks = (unsigned)((BATCH * DIN / 2 + 3) / 4); // 4 waves/blk
        selective_scan_kernel<<<nblocks, 128, 0, stream>>>(
            uf_bf, deltaf_bf, proj_f, A_log, Dv, xz, outz, 0);
        selective_scan_kernel<<<nblocks, 128, 0, stream>>>(
            ub_bf, deltab_bf, proj_b, A_b_log, Dv_b, xz, outz, 1);
    }

    // out = out_z @ W_out^T
    cvt(outz, outz_bf, (long)MTOT * DIN);
    gemm(outz_bf, DIN, Wout_bf, DIN, MTOT, DMOD, DIN, out, DMOD,
         nullptr, 0, 0, nullptr);
}
